// Decode_91164975825419
// MI455X (gfx1250) — matrix operand order repaired and checked by reading, not compiled, not run on hardware
//
#include <hip/hip_runtime.h>
#include <hip/hip_bf16.h>
#include <math.h>

typedef _Float16 h16;
typedef __attribute__((ext_vector_type(16))) _Float16 v16h;
typedef __attribute__((ext_vector_type(2)))  _Float16 v2h;
typedef __attribute__((ext_vector_type(8)))  float    v8f;

#define HID   256
#define FEAT  128
#define BATCH 32768

#define AM  32      // LSTM kernel rows per block (2 waves)
#define ATH 64
#define BM  64      // MLP kernel rows per block (4 waves)
#define BTH 128

#if __has_builtin(__builtin_amdgcn_sched_barrier)
#define SCHED_FENCE() __builtin_amdgcn_sched_barrier(0)
#else
#define SCHED_FENCE()
#endif

// ---------------- packed-weight layout ----------------
// B-fragment tile = 32(K) x 16(N) f16, 512 halves, stored [lane][half]:
//   lane l: col = l&15, kbase = (l>>4)*8 ; half j -> K = kbase + (j<8 ? j : j+8)
#define OFF_WX   0u        // 256x512 -> 131072 halves
#define OFF_UH   131072u   // 128x512 ->  65536
#define OFF_WXT  196608u   // 256x128 ->  32768
#define OFF_W1   229376u   // 128x128 ->  16384
#define OFF_W2   245760u
#define OFF_W3   262144u   // end = 278528 halves = 557056 bytes
#define OFF_HSEQ_BYTES 557056u   // h16 hseq[B*3][128] = 25165824 bytes

__global__ void pack_b_kernel(const float* __restrict__ W, h16* __restrict__ out,
                              int K, int N) {
  int idx = blockIdx.x * 256 + threadIdx.x;
  int total = K * N;
  if (idx >= total) return;
  int tile   = idx >> 9;
  int within = idx & 511;
  int lanei  = within >> 4;
  int j      = within & 15;
  int ntiles = N >> 4;
  int kt = tile / ntiles;
  int nt = tile - kt * ntiles;
  int col   = lanei & 15;
  int kbase = (lanei >> 4) << 3;
  int Kg = kt * 32 + kbase + (j < 8 ? j : j + 8);
  int Ng = nt * 16 + col;
  out[idx] = (h16)W[(size_t)Kg * N + Ng];
}

// ---------------- helpers ----------------
__device__ __forceinline__ float sigf(float x) { return 1.0f / (1.0f + __expf(-x)); }

__device__ __forceinline__ v8f wmma16(v16h a, v16h b, v8f c) {
  return __builtin_amdgcn_wmma_f32_16x16x32_f16(false, a, false, b, (short)0, c,
                                                false, false);
}

__device__ __forceinline__ v16h load_bfrag(const h16* __restrict__ base,
                                           int kf, int nt, int ntiles, int lane) {
  const h16* p = base + ((((size_t)kf * ntiles) + nt) << 9) + (lane << 4);
  return *(const v16h*)p;
}

// A-fragment stash in LDS: [frag][lane][16 halves] (32B per lane, aligned)
__device__ __forceinline__ v16h load_afrag(const h16* sw, int fbase, int kf, int lane) {
  return *(const v16h*)(sw + (((fbase + kf) << 5) + lane) * 16);
}
__device__ __forceinline__ void store_afrag(h16* sw, int fbase, int kf, int lane, v16h f) {
  *(v16h*)(sw + (((fbase + kf) << 5) + lane) * 16) = f;
}

// i/f/o/g pre-activation for one 16-col tile: A-frags stream from LDS,
// B-frags stream from global/L2.
__device__ __forceinline__ v8f gate_acc(const h16* xswz, const h16* hswz,
                                        int xbase, int hbase,
                                        const h16* __restrict__ pWx,
                                        const h16* __restrict__ pUh,
                                        int nt, int lane) {
  v8f a = v8f{};
#pragma unroll
  for (int kf = 0; kf < 8; ++kf)
    a = wmma16(load_afrag(xswz, xbase, kf, lane),
               load_bfrag(pWx, kf, nt, 32, lane), a);
#pragma unroll
  for (int kf = 0; kf < 4; ++kf)
    a = wmma16(load_afrag(hswz, hbase, kf, lane),
               load_bfrag(pUh, kf, nt, 32, lane), a);
  return a;
}

// ---------------- kernel A : fused time-LSTM (3 steps) ----------------
__global__ __launch_bounds__(ATH, 4)
void tlstm_kernel(const float* __restrict__ ctx,      // [B,3,HID]
                  const float* __restrict__ tin,      // [B,6]
                  const float* __restrict__ aw1,
                  const float* __restrict__ aw2,
                  const float* __restrict__ aw3,
                  const float* __restrict__ bias,     // [512]
                  const float* __restrict__ Wtt,      // [128]
                  const float* __restrict__ bt,       // [128]
                  const float* __restrict__ Wto,      // [128]
                  const h16*  __restrict__ wpk,       // packed f16 weights
                  h16* __restrict__ hseq)             // [B*3,FEAT] f16
{
  __shared__ __align__(32) h16   lh  [AM * HID];    // 16 KB : last_hidden (f16)
  __shared__ __align__(32) h16   hbuf[AM * FEAT];   //  8 KB : recurrent h  (f16)
  __shared__ __align__(32) float cbuf[AM * FEAT];   // 16 KB : cell state   (f32)
  __shared__ __align__(32) h16   xswz[2 * 8 * 32 * 16]; // 16 KB : x A-frags
  __shared__ __align__(32) h16   hswz[2 * 4 * 32 * 16]; //  8 KB : h A-frags

  const int tid  = threadIdx.x;
  const int lane = tid & 31;
  const int wave = tid >> 5;
  const int rowBase = blockIdx.x * AM;

  for (int i = tid; i < AM * HID; i += ATH) {
    int r = i >> 8;
    int k = i & (HID - 1);
    lh[i] = (h16)ctx[(size_t)(rowBase + r) * 3 * HID + 2 * HID + k];
  }
  for (int i = tid; i < AM * FEAT; i += ATH) { hbuf[i] = (h16)0.0f; cbuf[i] = 0.0f; }
  __syncthreads();

  const int r    = lane & 15;            // A-frag row within wave tile
  const int kb   = (lane < 16) ? 0 : 8;  // A-frag K base select
  const int mOff = (lane >> 4) << 3;     // C/D layout row offset (0 or 8)
  const int nloc = lane & 15;            // C/D layout column
  const int wrow = wave * 16;
  const int xbase = wave * 8;            // frag-region bases in xswz/hswz
  const int hbase = wave * 4;

  const h16* pWx  = wpk + OFF_WX;
  const h16* pUh  = wpk + OFF_UH;
  const h16* pWxt = wpk + OFF_WXT;

#pragma unroll 1
  for (int s = 0; s < 3; ++s) {
    const float* aws = (s == 0) ? aw1 : ((s == 1) ? aw2 : aw3);

    float tv[8];
#pragma unroll
    for (int v = 0; v < 8; ++v)
      tv[v] = tin[(size_t)(rowBase + wrow + mOff + v) * 6 + 3 + s];

    // build x A-fragments (x = last_hidden * attn[s]) and stash to LDS
#pragma unroll
    for (int kf = 0; kf < 8; ++kf) {
      v16h f;
#pragma unroll
      for (int v = 0; v < 8; ++v) {
        int j0 = 2 * v;
        int K  = kf * 32 + kb + (j0 < 8 ? j0 : j0 + 8);
        v2h p  = *(const v2h*)(lh + (wrow + r) * HID + K);
        f[2 * v]     = (h16)((float)p.x * aws[K]);
        f[2 * v + 1] = (h16)((float)p.y * aws[K + 1]);
      }
      store_afrag(xswz, xbase, kf, lane, f);
    }
    // build h A-fragments and stash to LDS
#pragma unroll
    for (int kf = 0; kf < 4; ++kf) {
      v16h f;
#pragma unroll
      for (int v = 0; v < 8; ++v) {
        int j0 = 2 * v;
        int K  = kf * 32 + kb + (j0 < 8 ? j0 : j0 + 8);
        v2h p  = *(const v2h*)(hbuf + (wrow + r) * FEAT + K);
        f[2 * v]     = p.x;
        f[2 * v + 1] = p.y;
      }
      store_afrag(hswz, hbase, kf, lane, f);
    }
    SCHED_FENCE();

#pragma unroll 1
    for (int cc = 0; cc < 8; ++cc) {
      v8f ai = gate_acc(xswz, hswz, xbase, hbase, pWx, pUh, 0 * 8 + cc, lane);
      SCHED_FENCE();
      v8f af = gate_acc(xswz, hswz, xbase, hbase, pWx, pUh, 1 * 8 + cc, lane);
      SCHED_FENCE();
      v8f ao = gate_acc(xswz, hswz, xbase, hbase, pWx, pUh, 2 * 8 + cc, lane);
      SCHED_FENCE();
      v8f ag = gate_acc(xswz, hswz, xbase, hbase, pWx, pUh, 3 * 8 + cc, lane);
      SCHED_FENCE();
      v8f aT = v8f{};
#pragma unroll
      for (int kf = 0; kf < 8; ++kf)
        aT = wmma16(load_afrag(xswz, xbase, kf, lane),
                    load_bfrag(pWxt, kf, cc, 8, lane), aT);
      SCHED_FENCE();

      int col = cc * 16 + nloc;
      float bi = bias[col], bf = bias[128 + col];
      float bo = bias[256 + col], bg = bias[384 + col];
      float wtt = Wtt[col], btc = bt[col], wto = Wto[col];
#pragma unroll
      for (int v = 0; v < 8; ++v) {
        int ridx = wrow + mOff + v;
        float iv = sigf(ai[v] + bi);
        float fv = sigf(af[v] + bf);
        float ov = sigf(ao[v] + bo + tv[v] * wto);
        float gv = tanhf(ag[v] + bg);
        float Tg = sigf(aT[v] + sigf(tv[v] * wtt) + btc);
        float cn = fv * cbuf[ridx * FEAT + col] + iv * Tg * gv;
        cbuf[ridx * FEAT + col] = cn;
        h16 hn = (h16)(ov * tanhf(cn));
        hbuf[ridx * FEAT + col] = hn;
        hseq[((size_t)(rowBase + ridx) * 3 + s) * FEAT + col] = hn;
      }
      SCHED_FENCE();
    }
    __syncthreads();
  }
}

// ---------------- kernel B : 3-layer ReLU MLP ----------------
__global__ __launch_bounds__(BTH, 4)
void mlp_kernel(const h16* __restrict__ hseq,   // [B*3,FEAT]
                const h16* __restrict__ wpk,
                const float* __restrict__ b1,
                const float* __restrict__ b2,
                const float* __restrict__ b3,
                float* __restrict__ out)         // [B*3,FEAT]
{
  __shared__ __align__(32) h16 buf[BM * FEAT];   // 16 KB, in-place layer buffer

  const int tid  = threadIdx.x;
  const int lane = tid & 31;
  const int wave = tid >> 5;
  const int rowBase = blockIdx.x * BM;

  // contiguous tile copy (f16 rows are contiguous in hseq)
  {
    const uint32_t* src = (const uint32_t*)(hseq + (size_t)rowBase * FEAT);
    uint32_t* dst = (uint32_t*)buf;
    for (int i = tid; i < BM * FEAT / 2; i += BTH) dst[i] = src[i];
  }
  __syncthreads();

  const int r    = lane & 15;
  const int kb   = (lane < 16) ? 0 : 8;
  const int mOff = (lane >> 4) << 3;
  const int nloc = lane & 15;
  const int wrow = wave * 16;

#pragma unroll 1
  for (int l = 0; l < 3; ++l) {
    const h16*  pW = (l == 0) ? (wpk + OFF_W1)
                  : (l == 1) ? (wpk + OFF_W2) : (wpk + OFF_W3);
    const float* pb = (l == 0) ? b1 : (l == 1) ? b2 : b3;

    v16h fa[4];                       // all reads before in-place writes
#pragma unroll
    for (int kf = 0; kf < 4; ++kf) {
      v16h f;
#pragma unroll
      for (int v = 0; v < 8; ++v) {
        int j0 = 2 * v;
        int K  = kf * 32 + kb + (j0 < 8 ? j0 : j0 + 8);
        v2h p  = *(const v2h*)(buf + (wrow + r) * FEAT + K);
        f[2 * v]     = p.x;
        f[2 * v + 1] = p.y;
      }
      fa[kf] = f;
    }
    SCHED_FENCE();

#pragma unroll 1
    for (int nt = 0; nt < 8; ++nt) {
      v8f a = v8f{};
#pragma unroll
      for (int kf = 0; kf < 4; ++kf)
        a = wmma16(fa[kf], load_bfrag(pW, kf, nt, 8, lane), a);
      int col = nt * 16 + nloc;
      float bb = pb[col];
#pragma unroll
      for (int v = 0; v < 8; ++v) {
        float x = a[v] + bb;
        x = x > 0.0f ? x : 0.0f;
        if (l < 2)
          buf[(wrow + mOff + v) * FEAT + col] = (h16)x;
        else
          out[(size_t)(rowBase + wrow + mOff + v) * FEAT + col] = x;
      }
      SCHED_FENCE();
    }
  }
}

// ---------------- host launcher ----------------
extern "C" void kernel_launch(void* const* d_in, const int* in_sizes, int n_in,
                              void* d_out, int out_size, void* d_ws, size_t ws_size,
                              hipStream_t stream) {
  const float* ctx = (const float*)d_in[0];
  const float* tin = (const float*)d_in[1];
  const float* aw1 = (const float*)d_in[2];
  const float* aw2 = (const float*)d_in[3];
  const float* aw3 = (const float*)d_in[4];
  const float* Wx  = (const float*)d_in[5];
  const float* Uh  = (const float*)d_in[6];
  const float* b   = (const float*)d_in[7];
  const float* Wxt = (const float*)d_in[8];
  const float* Wtt = (const float*)d_in[9];
  const float* bt  = (const float*)d_in[10];
  const float* Wto = (const float*)d_in[11];
  const float* w1  = (const float*)d_in[12];
  const float* b1  = (const float*)d_in[13];
  const float* w2  = (const float*)d_in[14];
  const float* b2  = (const float*)d_in[15];
  const float* w3  = (const float*)d_in[16];
  const float* b3  = (const float*)d_in[17];

  h16* wpk  = (h16*)d_ws;
  h16* hseq = (h16*)((char*)d_ws + OFF_HSEQ_BYTES);

  pack_b_kernel<<<(256 * 512 + 255) / 256, 256, 0, stream>>>(Wx,  wpk + OFF_WX,  256, 512);
  pack_b_kernel<<<(128 * 512 + 255) / 256, 256, 0, stream>>>(Uh,  wpk + OFF_UH,  128, 512);
  pack_b_kernel<<<(256 * 128 + 255) / 256, 256, 0, stream>>>(Wxt, wpk + OFF_WXT, 256, 128);
  pack_b_kernel<<<(128 * 128 + 255) / 256, 256, 0, stream>>>(w1,  wpk + OFF_W1,  128, 128);
  pack_b_kernel<<<(128 * 128 + 255) / 256, 256, 0, stream>>>(w2,  wpk + OFF_W2,  128, 128);
  pack_b_kernel<<<(128 * 128 + 255) / 256, 256, 0, stream>>>(w3,  wpk + OFF_W3,  128, 128);

  tlstm_kernel<<<BATCH / AM, ATH, 0, stream>>>(
      ctx, tin, aw1, aw2, aw3, b, Wtt, bt, Wto, wpk, hseq);

  mlp_kernel<<<(BATCH * 3) / BM, BTH, 0, stream>>>(
      hseq, wpk, b1, b2, b3, (float*)d_out);
}
